// FlashRWAttention_51410758533418
// MI455X (gfx1250) — matrix operand order repaired and checked by reading, not compiled, not run on hardware
//
#include <hip/hip_runtime.h>

// ---------------------------------------------------------------------------
// Problem constants (match reference)
// ---------------------------------------------------------------------------
#define HH   32
#define DD   128
#define HID  4096          // HH*DD
#define BB   4
#define SS   1024
#define TT   4096          // BB*SS
#define NQKV 4352          // HID + 2*DD
#define NSLOTS 16384

typedef __attribute__((ext_vector_type(16))) __bf16 v16bf;
typedef __attribute__((ext_vector_type(8)))  float  v8f;

static __device__ inline v8f vzero8() {
  v8f z;
#pragma unroll
  for (int i = 0; i < 8; ++i) z[i] = 0.0f;
  return z;
}

// Build a 16-element bf16 fragment from two 16-byte-aligned 8-element chunks.
static __device__ inline v16bf ldfrag(const __bf16* p0, const __bf16* p1) {
  v16bf r;
  __builtin_memcpy(&r, p0, 16);
  __builtin_memcpy(((char*)&r) + 16, p1, 16);
  return r;
}

static __device__ inline v8f wmma_bf16(v16bf a, v16bf b, v8f c) {
  // D = A(16x32) * B(32x16) + C, fp32 accumulate
  return __builtin_amdgcn_wmma_f32_16x16x32_bf16(false, a, false, b,
                                                 (short)0, c, false, false);
}

// CDNA5 async memory->LDS copy (16B per lane), tracked by ASYNCcnt.
// LDS address operand = low 32 bits of the flat address (aperture rule).
static __device__ inline void async_copy_b128(void* lds_ptr, const void* gptr) {
  unsigned lds_off = (unsigned)(uintptr_t)lds_ptr;
  asm volatile("global_load_async_to_lds_b128 %0, %1, off"
               :: "v"(lds_off), "v"((unsigned long long)(uintptr_t)gptr)
               : "memory");
}
static __device__ inline void wait_async0() {
  asm volatile("s_wait_asynccnt 0x0" ::: "memory");
}

// ---------------------------------------------------------------------------
// Tiled GEMM: C(MxN,f32) = A(MxK,bf16) * B(KxN,f32->bf16) + bias(N)
// Block 256 threads = 8 waves; BM=BN=128, BK=32; wave tile 64x32 (4x2 WMMAs).
// Double-buffered LDS; A tile staged via GLOBAL_LOAD_ASYNC_TO_LDS_B128
// overlapping the previous tile's WMMAs; one barrier per k-step.
// ---------------------------------------------------------------------------
__global__ __launch_bounds__(256)
void gemm_wmma_kernel(const __bf16* __restrict__ A,
                      const float* __restrict__ B,
                      const float* __restrict__ bias,
                      float* __restrict__ C,
                      int M, int N, int K) {
  __shared__ __align__(16) __bf16 Al[2][128][40];  // padded rows: 80B (16B mult)
  __shared__ __align__(16) __bf16 Bt[2][128][40];  // B transposed: [n][k]

  const int tid  = threadIdx.x;
  const int lane = tid & 31;
  const int wv   = tid >> 5;
  const int wm   = wv >> 2;     // 0..1  (64 rows each)
  const int wn   = wv & 3;      // 0..3  (32 cols each)
  const int lm   = lane & 15;
  const int hi   = lane >> 4;

  const int row0 = blockIdx.y * 128;
  const int col0 = blockIdx.x * 128;

  v8f acc[4][2];
#pragma unroll
  for (int ms = 0; ms < 4; ++ms)
#pragma unroll
    for (int ns = 0; ns < 2; ++ns) acc[ms][ns] = vzero8();

  // ---- tile staging: async A copy + f32->bf16 transposed B ----
  auto stage = [&](int kt, int buf) {
#pragma unroll
    for (int r = 0; r < 2; ++r) {                  // A: 512 x 16B chunks
      int idx = tid + r * 256;
      int m = idx >> 2, c8 = (idx & 3) * 8;
      async_copy_b128(&Al[buf][m][c8],
                      A + (size_t)(row0 + m) * K + kt + c8);
    }
#pragma unroll
    for (int r = 0; r < 4; ++r) {                  // B: 1024 float4s
      int idx = tid + r * 256;
      int k = idx >> 5, n4 = (idx & 31) * 4;
      float4 v = *(const float4*)(B + (size_t)(kt + k) * N + col0 + n4);
      Bt[buf][n4 + 0][k] = (__bf16)v.x; Bt[buf][n4 + 1][k] = (__bf16)v.y;
      Bt[buf][n4 + 2][k] = (__bf16)v.z; Bt[buf][n4 + 3][k] = (__bf16)v.w;
    }
    __builtin_prefetch(B + (size_t)(kt + 32) * N + col0 + (tid & 31) * 4, 0, 1);
  };

  stage(0, 0);
  for (int kt = 0; kt < K; kt += 32) {
    const int cur = (kt >> 5) & 1;
    wait_async0();          // this wave's tile-kt async data is in LDS
    __syncthreads();        // all waves' tile-kt data visible; frees buf^1
    if (kt + 32 < K) stage(kt + 32, cur ^ 1);   // overlaps compute below

    v16bf af[4], bf_[2];
#pragma unroll
    for (int ms = 0; ms < 4; ++ms) {
      int row = wm * 64 + ms * 16 + lm;
      af[ms] = ldfrag(&Al[cur][row][8 * hi], &Al[cur][row][16 + 8 * hi]);
    }
#pragma unroll
    for (int ns = 0; ns < 2; ++ns) {
      int col = wn * 32 + ns * 16 + lm;
      bf_[ns] = ldfrag(&Bt[cur][col][8 * hi], &Bt[cur][col][16 + 8 * hi]);
    }
#pragma unroll
    for (int ms = 0; ms < 4; ++ms)
#pragma unroll
      for (int ns = 0; ns < 2; ++ns)
        acc[ms][ns] = wmma_bf16(af[ms], bf_[ns], acc[ms][ns]);
  }

  // ---- epilogue: bias add, f32 store ----
#pragma unroll
  for (int ms = 0; ms < 4; ++ms)
#pragma unroll
    for (int ns = 0; ns < 2; ++ns) {
      int gcol = col0 + wn * 32 + ns * 16 + lm;
      float bv = bias[gcol];
#pragma unroll
      for (int r = 0; r < 8; ++r) {
        int grow = row0 + wm * 64 + ms * 16 + r + 8 * hi;
        C[(size_t)grow * N + gcol] = acc[ms][ns][r] + bv;
      }
    }
}

// ---------------------------------------------------------------------------
// One-shot f32 -> bf16 conversion (hidden_states), memory bound.
// ---------------------------------------------------------------------------
__global__ __launch_bounds__(256)
void cvt_f32_bf16_kernel(const float* __restrict__ in,
                         __bf16* __restrict__ out, int n4) {
  int i = blockIdx.x * blockDim.x + threadIdx.x;
  int stride = gridDim.x * blockDim.x;
  for (; i < n4; i += stride) {
    float4 v = *(const float4*)(in + (size_t)i * 4);
    __bf16* o = out + (size_t)i * 4;
    o[0] = (__bf16)v.x; o[1] = (__bf16)v.y; o[2] = (__bf16)v.z; o[3] = (__bf16)v.w;
  }
}

// ---------------------------------------------------------------------------
// RoPE + bf16 conversion + cache scatter. One block (128 thr) per token.
// Produces q_bf16[T][HID], k_bf16[T][D], V^T bf16 [B][D][S]; fills caches.
// ---------------------------------------------------------------------------
__global__ __launch_bounds__(128)
void rope_scatter_kernel(const float* __restrict__ qkv,
                         const float* __restrict__ cosp,
                         const float* __restrict__ sinp,
                         const int* __restrict__ slots,
                         __bf16* __restrict__ qbf,
                         __bf16* __restrict__ kbf,
                         __bf16* __restrict__ vtbf,
                         float* __restrict__ kcache,
                         float* __restrict__ vcache) {
  const int t   = blockIdx.x;
  const int tid = threadIdx.x;        // 0..127
  const int j   = tid & 63;
  const int sel = tid >> 6;
  const float c = cosp[t * 64 + j];
  const float s = sinp[t * 64 + j];
  const float* row = qkv + (size_t)t * NQKV;
  const int slot = slots[t];

#pragma unroll 4
  for (int h = 0; h < HH; ++h) {
    float x1 = row[h * DD + j];
    float x2 = row[h * DD + 64 + j];
    float o  = sel ? (x2 * c + x1 * s) : (x1 * c - x2 * s);
    qbf[(size_t)t * HID + h * DD + sel * 64 + j] = (__bf16)o;
  }
  { // K (rope + scatter)
    float x1 = row[HID + j];
    float x2 = row[HID + 64 + j];
    float o  = sel ? (x2 * c + x1 * s) : (x1 * c - x2 * s);
    int d = sel * 64 + j;
    kbf[(size_t)t * DD + d] = (__bf16)o;
    kcache[(size_t)slot * DD + d] = o;
  }
  { // V (scatter + transpose)
    float v = row[HID + DD + tid];
    vcache[(size_t)slot * DD + tid] = v;
    int b = t >> 10, sidx = t & 1023;
    vtbf[((size_t)b * DD + tid) * SS + sidx] = (__bf16)v;
  }
}

// ---------------------------------------------------------------------------
// Flash MQA attention. Block = 128 thr (4 waves); block owns (b, h, 64 q rows),
// each wave owns 16 q rows. Online softmax; 32-key KV groups;
// QK^T: 2x4 WMMAs, PV: 8 WMMAs per group. K/V are L2-resident (MQA).
// ---------------------------------------------------------------------------
__global__ __launch_bounds__(128)
void attn_kernel(const __bf16* __restrict__ qbf,
                 const __bf16* __restrict__ kbf,
                 const __bf16* __restrict__ vtbf,
                 __bf16* __restrict__ attnbf) {
  __shared__ __align__(16) __bf16 Pl[4][16][32];   // per-wave P tile

  const int bid = blockIdx.x;
  const int qt  = bid & 15;
  const int h   = (bid >> 4) & 31;
  const int b   = bid >> 9;
  const int wv  = threadIdx.x >> 5;
  const int lane = threadIdx.x & 31;
  const int lm  = lane & 15;
  const int hi  = lane >> 4;
  const int q0  = qt * 64 + wv * 16;              // within-batch query base
  const float scale = 0.08838834764831845f;       // 128^-0.5

  // preload Q fragments (16 rows x 128 dims = 4 k-chunks)
  v16bf qf[4];
  {
    const __bf16* qb = qbf + (size_t)(b * SS + q0 + lm) * HID + h * DD;
#pragma unroll
    for (int jc = 0; jc < 4; ++jc)
      qf[jc] = ldfrag(qb + jc * 32 + 8 * hi, qb + jc * 32 + 16 + 8 * hi);
  }

  v8f o[8];
#pragma unroll
  for (int dc = 0; dc < 8; ++dc) o[dc] = vzero8();
  float rowmax[8], lsum[8];
#pragma unroll
  for (int r = 0; r < 8; ++r) { rowmax[r] = -1e30f; lsum[r] = 0.0f; }

  for (int kb = 0; kb <= q0 + 15; kb += 32) {
#pragma unroll
    for (int t2 = 0; t2 < 2; ++t2) {
      const int keyb = kb + t2 * 16;
      if (keyb <= q0 + 15) {
        v8f sc = vzero8();
        const __bf16* kbase = kbf + (size_t)(b * SS + keyb + lm) * DD;
#pragma unroll
        for (int jc = 0; jc < 4; ++jc) {
          v16bf kf = ldfrag(kbase + jc * 32 + 8 * hi,
                            kbase + jc * 32 + 16 + 8 * hi);
          sc = wmma_bf16(qf[jc], kf, sc);
        }
        float sv[8], tmax[8];
        const int krow = keyb + lm;
#pragma unroll
        for (int r = 0; r < 8; ++r) {
          int qrow = q0 + r + 8 * hi;
          float x = sc[r] * scale;
          sv[r] = (krow <= qrow) ? x : -1e30f;
          tmax[r] = sv[r];
        }
#pragma unroll
        for (int m = 1; m < 16; m <<= 1)
#pragma unroll
          for (int r = 0; r < 8; ++r)
            tmax[r] = fmaxf(tmax[r], __shfl_xor(tmax[r], m, 32));
#pragma unroll
        for (int r = 0; r < 8; ++r) {
          float nm = fmaxf(rowmax[r], tmax[r]);
          float al = __expf(rowmax[r] - nm);
          rowmax[r] = nm;
          lsum[r] *= al;
#pragma unroll
          for (int dc = 0; dc < 8; ++dc) o[dc][r] *= al;
        }
#pragma unroll
        for (int r = 0; r < 8; ++r) {
          float p = __expf(sv[r] - rowmax[r]);
          lsum[r] += p;
          Pl[wv][r + 8 * hi][t2 * 16 + lm] = (__bf16)p;
        }
      } else {
#pragma unroll
        for (int r = 0; r < 8; ++r)
          Pl[wv][r + 8 * hi][t2 * 16 + lm] = (__bf16)0.0f;
      }
    }
    // P (16x32) as A-fragment; V^T gives contiguous B-fragments
    v16bf pf = ldfrag(&Pl[wv][lm][8 * hi], &Pl[wv][lm][16 + 8 * hi]);
#pragma unroll
    for (int dc = 0; dc < 8; ++dc) {
      const __bf16* vb = vtbf + ((size_t)(b * DD + dc * 16 + lm)) * SS + kb + 8 * hi;
      v16bf vf = ldfrag(vb, vb + 16);
      o[dc] = wmma_bf16(pf, vf, o[dc]);
    }
  }

  // normalize and store as bf16 for the dense GEMM
#pragma unroll
  for (int m = 1; m < 16; m <<= 1)
#pragma unroll
    for (int r = 0; r < 8; ++r) lsum[r] += __shfl_xor(lsum[r], m, 32);
#pragma unroll
  for (int r = 0; r < 8; ++r) {
    float rinv = 1.0f / lsum[r];
    size_t trow = (size_t)(b * SS + q0 + r + 8 * hi);
#pragma unroll
    for (int dc = 0; dc < 8; ++dc)
      attnbf[trow * HID + h * DD + dc * 16 + lm] = (__bf16)(o[dc][r] * rinv);
  }
}

// ---------------------------------------------------------------------------
// Launch
// ---------------------------------------------------------------------------
extern "C" void kernel_launch(void* const* d_in, const int* in_sizes, int n_in,
                              void* d_out, int out_size, void* d_ws, size_t ws_size,
                              hipStream_t stream) {
  (void)in_sizes; (void)n_in; (void)out_size; (void)ws_size;
  const float* hs    = (const float*)d_in[0];
  const float* cosp  = (const float*)d_in[1];
  const float* sinp  = (const float*)d_in[2];
  const float* kc_in = (const float*)d_in[3];
  const float* vc_in = (const float*)d_in[4];
  const float* qkvw  = (const float*)d_in[5];
  const float* qkvb  = (const float*)d_in[6];
  const float* dw    = (const float*)d_in[7];
  const float* db    = (const float*)d_in[8];
  const int*   slots = (const int*)d_in[9];

  float* out    = (float*)d_out;
  float* kcache = out + (size_t)TT * HID;
  float* vcache = kcache + (size_t)NSLOTS * DD;

  char* ws = (char*)d_ws;
  float*  qkvf   = (float*)ws;                          // 71,303,168 B (T x 4352 f32)
  __bf16* attnbf = (__bf16*)ws;                         // reuses qkvf region (dead by then)
  __bf16* qbf    = (__bf16*)(ws + 71303168);            // 33,554,432 B
  __bf16* kbf    = (__bf16*)(ws + 71303168 + 33554432);             // 1 MiB
  __bf16* vtbf   = (__bf16*)(ws + 71303168 + 33554432 + 1048576);   // 1 MiB
  __bf16* hsbf   = (__bf16*)(ws + 71303168 + 33554432 + 2097152);   // 33,554,432 B

  // 0) one-shot bf16 conversion of hidden_states (A of GEMM1)
  cvt_f32_bf16_kernel<<<2048, 256, 0, stream>>>(hs, hsbf, (TT * HID) / 4);

  // 1) QKV projection (+bias): async-LDS staged bf16 WMMA, fp32 out
  gemm_wmma_kernel<<<dim3(NQKV / 128, TT / 128), 256, 0, stream>>>(
      hsbf, qkvw, qkvb, qkvf, TT, NQKV, HID);

  // 2) initialize caches from inputs, then scatter
  hipMemcpyAsync(kcache, kc_in, (size_t)NSLOTS * DD * sizeof(float),
                 hipMemcpyDeviceToDevice, stream);
  hipMemcpyAsync(vcache, vc_in, (size_t)NSLOTS * DD * sizeof(float),
                 hipMemcpyDeviceToDevice, stream);
  rope_scatter_kernel<<<TT, 128, 0, stream>>>(qkvf, cosp, sinp, slots,
                                              qbf, kbf, vtbf, kcache, vcache);

  // 3) causal MQA flash attention (bf16 WMMA, fp32 softmax)
  attn_kernel<<<BB * HH * (SS / 64), 128, 0, stream>>>(qbf, kbf, vtbf, attnbf);

  // 4) dense projection (+bias) -> f32 output
  gemm_wmma_kernel<<<dim3(HID / 128, TT / 128), 256, 0, stream>>>(
      attnbf, dw, db, out, TT, HID, HID);
}
